// ArrayMultiplier24x24_23407571764133
// MI455X (gfx1250) — matrix operand order repaired and checked by reading, not compiled, not run on hardware
//
#include <hip/hip_runtime.h>

// 24x24-bit array multiplier on 0/1 float bit-vectors (LSB first).
// Semantics: a = sum_i A[i]*2^i ; b = sum_j B[j]*2^j ; out bits = (a*b) bits 0..47.
//
// Strategy (MI455X / gfx1250, wave32):
//  - Each wave handles 16 rows.
//  - Bit->integer packing done on the matrix unit: V_WMMA_F32_16X16X32_BF16 with a
//    constant weight B-matrix W[k,n] = 2^k (exact in bf16; sums < 2^24 exact in f32).
//  - Each lane then owns one (row, 24-bit output half): exact u64 multiply + bit unpack,
//    one contiguous 96B store per lane.
// Memory-bound problem (~25 MB total traffic): loads/stores are 128-bit vectorized.

typedef __attribute__((ext_vector_type(16))) __bf16 v16bf;
typedef __attribute__((ext_vector_type(8)))  float  v8f;

__device__ __forceinline__ v16bf load_bits_frag(const float* __restrict__ row, int lane) {
    // CDNA5 16-bit A-matrix 16x32 layout:
    //   lanes 0-15 : M=lane,     elems 0..7 = K0..7,  elems 8..15 = K16..23
    //   lanes 16-31: M=lane-16,  elems 0..7 = K8..15, elems 8..15 = K24..31 (zero pad, only 24 bits)
    v16bf f;
    if (lane < 16) {
        float4 x0 = *(const float4*)(row + 0);
        float4 x1 = *(const float4*)(row + 4);
        float4 x2 = *(const float4*)(row + 16);
        float4 x3 = *(const float4*)(row + 20);
        f[0]  = (__bf16)x0.x; f[1]  = (__bf16)x0.y; f[2]  = (__bf16)x0.z; f[3]  = (__bf16)x0.w;
        f[4]  = (__bf16)x1.x; f[5]  = (__bf16)x1.y; f[6]  = (__bf16)x1.z; f[7]  = (__bf16)x1.w;
        f[8]  = (__bf16)x2.x; f[9]  = (__bf16)x2.y; f[10] = (__bf16)x2.z; f[11] = (__bf16)x2.w;
        f[12] = (__bf16)x3.x; f[13] = (__bf16)x3.y; f[14] = (__bf16)x3.z; f[15] = (__bf16)x3.w;
    } else {
        float4 x0 = *(const float4*)(row + 8);
        float4 x1 = *(const float4*)(row + 12);
        f[0]  = (__bf16)x0.x; f[1]  = (__bf16)x0.y; f[2]  = (__bf16)x0.z; f[3]  = (__bf16)x0.w;
        f[4]  = (__bf16)x1.x; f[5]  = (__bf16)x1.y; f[6]  = (__bf16)x1.z; f[7]  = (__bf16)x1.w;
        __bf16 z = (__bf16)0.0f;
        f[8] = z; f[9] = z; f[10] = z; f[11] = z; f[12] = z; f[13] = z; f[14] = z; f[15] = z;
    }
    return f;
}

__device__ __forceinline__ v16bf weight_frag(int lane) {
    // B-matrix 32x16 layout (matches A): lanes 0-15 elems 0..7 = K0..7, 8..15 = K16..23;
    // lanes 16-31 elems 0..7 = K8..15, 8..15 = K24..31. Column-constant weights: W[k,n] = 2^k.
    // bf16(2^k) bit pattern = (127+k) << 7 (pure exponent, exact).
    int kb = (lane < 16) ? 0 : 8;
    v16bf w;
#pragma unroll
    for (int j = 0; j < 16; ++j) {
        unsigned short u = (unsigned short)((127 + kb + ((j & 8) << 1) + (j & 7)) << 7);
        w[j] = __builtin_bit_cast(__bf16, u);
    }
    return w;
}

__global__ void __launch_bounds__(256)
ArrayMultiplier24x24_kernel(const float* __restrict__ A,
                            const float* __restrict__ B,
                            float* __restrict__ out,
                            int nrows) {
    const int tid  = threadIdx.x;
    const int wave = tid >> 5;   // 8 waves per block
    const int lane = tid & 31;

    const long long rowBase = (long long)blockIdx.x * 128 + (long long)wave * 16;
    if (rowBase >= nrows) return;  // wave-uniform; EXEC stays all-1s for active waves

    const int m = lane & 15;
    const float* aRow = A + (rowBase + m) * 24;
    const float* bRow = B + (rowBase + m) * 24;

    v16bf w  = weight_frag(lane);
    v16bf fa = load_bits_frag(aRow, lane);
    v16bf fb = load_bits_frag(bRow, lane);

    v8f ca = {0.f, 0.f, 0.f, 0.f, 0.f, 0.f, 0.f, 0.f};
    v8f cb = {0.f, 0.f, 0.f, 0.f, 0.f, 0.f, 0.f, 0.f};
    // a_int / b_int for 16 rows (replicated across the 16 columns of C)
    ca = __builtin_amdgcn_wmma_f32_16x16x32_bf16(false, fa, false, w, (short)0, ca, false, false);
    cb = __builtin_amdgcn_wmma_f32_16x16x32_bf16(false, fb, false, w, (short)0, cb, false, false);

    // C layout: lane<16 -> VGPR r = row r; lane>=16 -> VGPR r = row 8+r (any column).
    // Lane L handles row r = L>>1, output half h = L&1; its row lives locally at
    // index li = (L&15)>>1 in ca/cb. No cross-lane traffic needed.
    const int li = (lane & 15) >> 1;
    const int h  = lane & 1;
    float afv = 0.f, bfv = 0.f;
#pragma unroll
    for (int rI = 0; rI < 8; ++rI) {
        if (li == rI) { afv = ca[rI]; bfv = cb[rI]; }
    }

    const unsigned int ai = (unsigned int)afv;   // exact: integer < 2^24 in f32
    const unsigned int bi = (unsigned int)bfv;
    const unsigned long long p = (unsigned long long)ai * (unsigned long long)bi;
    const unsigned int bits = (unsigned int)(p >> (24 * h));  // low or high 24 product bits

    const int r = lane >> 1;
    float* orow = out + (rowBase + r) * 48 + h * 24;   // 96B contiguous per lane, 16B aligned
#pragma unroll
    for (int t = 0; t < 24; t += 4) {
        float4 v;
        v.x = (float)((bits >> (t + 0)) & 1u);
        v.y = (float)((bits >> (t + 1)) & 1u);
        v.z = (float)((bits >> (t + 2)) & 1u);
        v.w = (float)((bits >> (t + 3)) & 1u);
        *(float4*)(orow + t) = v;
    }
}

extern "C" void kernel_launch(void* const* d_in, const int* in_sizes, int n_in,
                              void* d_out, int out_size, void* d_ws, size_t ws_size,
                              hipStream_t stream) {
    const float* A = (const float*)d_in[0];
    const float* B = (const float*)d_in[1];
    float* out = (float*)d_out;

    const int nrows = in_sizes[0] / 24;            // 65536
    const int rowsPerBlock = 128;                  // 8 waves * 16 rows
    const int blocks = (nrows + rowsPerBlock - 1) / rowsPerBlock;  // 512

    ArrayMultiplier24x24_kernel<<<blocks, 256, 0, stream>>>(A, B, out, nrows);
}
